// MultiHeadSelfAttention_52939766890727
// MI455X (gfx1250) — matrix operand order repaired and checked by reading, not compiled, not run on hardware
//
#include <hip/hip_runtime.h>
#include <hip/hip_bf16.h>

typedef __attribute__((ext_vector_type(16))) _Float16 v16h;
typedef __attribute__((ext_vector_type(8)))  float    v8f;

#define Bb   8
#define Nn   1024
#define Dd   768
#define Hh   12
#define PD   64
#define Mtot (Bb * Nn)   /* 8192 */

#if defined(__HIP_DEVICE_COMPILE__) && __has_builtin(__builtin_amdgcn_global_load_async_to_lds_b128)
#define HAVE_ASYNC_LDS 1
#else
#define HAVE_ASYNC_LDS 0
#endif
#if defined(__HIP_DEVICE_COMPILE__) && __has_builtin(__builtin_amdgcn_s_wait_asynccnt)
#define HAVE_WAIT_ASYNC 1
#else
#define HAVE_WAIT_ASYNC 0
#endif

#if HAVE_ASYNC_LDS
// exact pointee type from the hipcc diagnostic: vector_size(16) int
typedef int v4i_vs __attribute__((vector_size(16)));
typedef __attribute__((address_space(1))) v4i_vs gvi4;   // global (AS1)
typedef __attribute__((address_space(3))) v4i_vs lvi4;   // LDS (AS3)
#endif

union V16H {
    v16h  v;
    uint4 u[2];
};

template <int N>
__device__ __forceinline__ void wait_async_le() {
#if HAVE_ASYNC_LDS
#if HAVE_WAIT_ASYNC
    __builtin_amdgcn_s_wait_asynccnt(N);
#else
    asm volatile("s_wait_asynccnt %0" :: "n"(N) : "memory");
#endif
#endif
}

// ---------------------------------------------------------------------------
// f32 -> f16 elementwise conversion
// ---------------------------------------------------------------------------
__global__ __launch_bounds__(256) void cvt_f16_kernel(const float* __restrict__ in,
                                                      _Float16* __restrict__ out, int n) {
    int i = blockIdx.x * 256 + threadIdx.x;
    if (i < n) out[i] = (_Float16)in[i];
}

// ---------------------------------------------------------------------------
// weight f32 [k][n] -> f16 transposed [n][k]  (dim = 768)
// ---------------------------------------------------------------------------
__global__ __launch_bounds__(256) void transpose_w_f16_kernel(const float* __restrict__ w,
                                                              _Float16* __restrict__ wt) {
    int i = blockIdx.x * 256 + threadIdx.x;       // over 768*768, write-coalesced
    int n = i / Dd;
    int k = i - n * Dd;
    wt[i] = (_Float16)w[k * Dd + n];
}

// ---------------------------------------------------------------------------
// Tiled WMMA GEMM:  C[M=8192][768] = A[M][768](f16) @ B (given as Bt[768][768] = B^T, f16)
// block tile 128x128x32, 256 threads = 8 waves (2 x 4), wave tile 64x32.
// Double-buffered LDS; staging via GLOBAL_LOAD_ASYNC_TO_LDS_B128 (ASYNCcnt).
// store modes: 0 = f16 head-split [B,H,N,64] (Q/K, Q pre-scaled by 0.125)
//              1 = f16 head-split transposed [B,H,64,N] (V)
//              2 = f32 flat [M][768] (final projection -> d_out)
// ---------------------------------------------------------------------------
__global__ __launch_bounds__(256) void gemm_wmma_kernel(
    const _Float16* __restrict__ A,
    const _Float16* __restrict__ Bt,
    const float*    __restrict__ bias,
    void*           __restrict__ outp,
    int mode, float scale)
{
    __shared__ _Float16 As[2][128][32];
    __shared__ _Float16 Bs[2][128][32];

    const int tid     = threadIdx.x;
    const int lane    = tid & 31;
    const int wave    = tid >> 5;
    const int lane_lo = lane & 15;
    const int lane_hi = lane >> 4;
    const int wm      = wave & 1;   // 0..1  -> M offset wm*64
    const int wn      = wave >> 1;  // 0..3  -> N offset wn*32
    const int gm0     = blockIdx.y * 128;
    const int gn0     = blockIdx.x * 128;

    const int lrow = tid >> 1;        // 0..127
    const int lcol = (tid & 1) * 16;  // 0 or 16

    const _Float16* gaBase = A  + (size_t)(gm0 + lrow) * Dd + lcol;
    const _Float16* gbBase = Bt + (size_t)(gn0 + lrow) * Dd + lcol;

    // stage one 128x32 A tile + 128x32 Bt tile into LDS buffer `buf`
    auto stage = [&](int buf, int kk) {
        const _Float16* ga = gaBase + kk;
        const _Float16* gb = gbBase + kk;
#if HAVE_ASYNC_LDS
        __builtin_amdgcn_global_load_async_to_lds_b128(
            (gvi4*)ga,       (lvi4*)&As[buf][lrow][lcol],     0, 0);
        __builtin_amdgcn_global_load_async_to_lds_b128(
            (gvi4*)(ga + 8), (lvi4*)&As[buf][lrow][lcol + 8], 0, 0);
        __builtin_amdgcn_global_load_async_to_lds_b128(
            (gvi4*)gb,       (lvi4*)&Bs[buf][lrow][lcol],     0, 0);
        __builtin_amdgcn_global_load_async_to_lds_b128(
            (gvi4*)(gb + 8), (lvi4*)&Bs[buf][lrow][lcol + 8], 0, 0);
#else
        uint4 a0 = *(const uint4*)ga, a1 = *(const uint4*)(ga + 8);
        uint4 b0 = *(const uint4*)gb, b1 = *(const uint4*)(gb + 8);
        *(uint4*)&As[buf][lrow][lcol]     = a0;
        *(uint4*)&As[buf][lrow][lcol + 8] = a1;
        *(uint4*)&Bs[buf][lrow][lcol]     = b0;
        *(uint4*)&Bs[buf][lrow][lcol + 8] = b1;
#endif
    };

    v8f acc[4][2] = {};

    constexpr int KT = Dd / 32;   // 24 k-tiles
    stage(0, 0);

    for (int kt = 0; kt < KT; ++kt) {
        const int cur = kt & 1;
        if (kt + 1 < KT) {
            stage(cur ^ 1, (kt + 1) * 32);
            wait_async_le<4>();   // oldest 4 async ops (current tile) have landed
        } else {
            wait_async_le<0>();
        }
        __syncthreads();

        V16H a[4], b[2];
        #pragma unroll
        for (int i = 0; i < 4; ++i) {
            const int m = wm * 64 + i * 16 + lane_lo;
            // A operand layout: e<8 -> k = 8*lane_hi + e ; e>=8 -> k = 16 + 8*lane_hi + (e-8)
            a[i].u[0] = *(const uint4*)&As[cur][m][lane_hi * 8];
            a[i].u[1] = *(const uint4*)&As[cur][m][16 + lane_hi * 8];
        }
        #pragma unroll
        for (int j = 0; j < 2; ++j) {
            const int n = wn * 32 + j * 16 + lane_lo;
            // B operand layout: e<8 -> k = 8*lane_hi + e ; e>=8 -> k = 16 + 8*lane_hi + (e-8)
            b[j].u[0] = *(const uint4*)&Bs[cur][n][lane_hi * 8];
            b[j].u[1] = *(const uint4*)&Bs[cur][n][16 + lane_hi * 8];
        }
        #pragma unroll
        for (int i = 0; i < 4; ++i)
            #pragma unroll
            for (int j = 0; j < 2; ++j)
                acc[i][j] = __builtin_amdgcn_wmma_f32_16x16x32_f16(
                    false, a[i].v, false, b[j].v, (short)0, acc[i][j], false, false);
        __syncthreads();
    }

    // store: C layout -> VGPR r holds row (8*lane_hi + r), column lane_lo
    #pragma unroll
    for (int i = 0; i < 4; ++i) {
        #pragma unroll
        for (int j = 0; j < 2; ++j) {
            const int n  = gn0 + wn * 32 + j * 16 + lane_lo;
            const float bn = bias[n];
            #pragma unroll
            for (int r = 0; r < 8; ++r) {
                const int m = gm0 + wm * 64 + i * 16 + lane_hi * 8 + r;
                const float c = (acc[i][j][r] + bn) * scale;
                if (mode == 2) {
                    ((float*)outp)[(size_t)m * Dd + n] = c;
                } else {
                    const int bidx = m >> 10, row = m & (Nn - 1);
                    const int h    = n >> 6,  d   = n & (PD - 1);
                    size_t idx;
                    if (mode == 0)
                        idx = (((size_t)(bidx * Hh + h)) * Nn + row) * PD + d;
                    else
                        idx = (((size_t)(bidx * Hh + h)) * PD + d) * Nn + row;
                    ((_Float16*)outp)[idx] = (_Float16)c;
                }
            }
        }
    }
}

// ---------------------------------------------------------------------------
// Flash attention: one wave = 16 q rows x full 64 d.  8 waves / block.
// Qh [BH][N][64], Kh [BH][N][64], Vt [BH][64][N]  (all f16, Q pre-scaled 1/8)
// attn out: f16 concat layout [B][N][768]
// ---------------------------------------------------------------------------
__global__ __launch_bounds__(256) void attention_kernel(
    const _Float16* __restrict__ Qh,
    const _Float16* __restrict__ Kh,
    const _Float16* __restrict__ Vt,
    _Float16*       __restrict__ attn)
{
    __shared__ _Float16 plds[8][16][32];   // per-wave P transpose scratch

    const int tid     = threadIdx.x;
    const int lane    = tid & 31;
    const int wave    = tid >> 5;
    const int lane_lo = lane & 15;
    const int lane_hi = lane >> 4;
    const int bh      = blockIdx.y;
    const int q0      = blockIdx.x * 128 + wave * 16;

    const _Float16* Qb = Qh + (size_t)bh * Nn * PD;
    const _Float16* Kb = Kh + (size_t)bh * Nn * PD;
    const _Float16* Vb = Vt + (size_t)bh * PD * Nn;

    // preload Q A-operands (loop invariant): d-chunks c = 0 (d 0..31), 1 (d 32..63)
    V16H qa[2];
    #pragma unroll
    for (int c = 0; c < 2; ++c) {
        const _Float16* p = Qb + (size_t)(q0 + lane_lo) * PD + c * 32 + lane_hi * 8;
        qa[c].u[0] = *(const uint4*)p;
        qa[c].u[1] = *(const uint4*)(p + 16);
    }

    v8f o[4] = {};                 // O accumulator: 16 q x 64 d (4 C-tiles)
    float mrun[8], lrun[8];
    #pragma unroll
    for (int r = 0; r < 8; ++r) { mrun[r] = -1e30f; lrun[r] = 0.0f; }

    for (int k0 = 0; k0 < Nn; k0 += 32) {
        // prefetch next key block (K rows and V^T rows) into cache
        if (k0 + 32 < Nn) {
            __builtin_prefetch(Kb + (size_t)(k0 + 32 + lane) * PD, 0, 0);
            __builtin_prefetch(Vb + (size_t)lane * Nn + k0 + 32, 0, 0);
            __builtin_prefetch(Vb + (size_t)(lane + 32) * Nn + k0 + 32, 0, 0);
        }

        // S = Q @ K^T for 32 keys -> two 16x16 f32 tiles
        v8f s[2] = {};
        #pragma unroll
        for (int j = 0; j < 2; ++j) {
            #pragma unroll
            for (int c = 0; c < 2; ++c) {
                V16H kb;
                const _Float16* p =
                    Kb + (size_t)(k0 + j * 16 + lane_lo) * PD + c * 32 + lane_hi * 8;
                kb.u[0] = *(const uint4*)p;
                kb.u[1] = *(const uint4*)(p + 16);
                s[j] = __builtin_amdgcn_wmma_f32_16x16x32_f16(
                    false, qa[c].v, false, kb.v, (short)0, s[j], false, false);
            }
        }

        // online softmax; lane covers q-row (8*lane_hi + r) for r = 0..7
        #pragma unroll
        for (int r = 0; r < 8; ++r) {
            const float v0 = s[0][r], v1 = s[1][r];
            float mx = fmaxf(v0, v1);
            #pragma unroll
            for (int off = 1; off < 16; off <<= 1)
                mx = fmaxf(mx, __shfl_xor(mx, off, 32));
            const float mnew  = fmaxf(mrun[r], mx);
            const float alpha = __expf(mrun[r] - mnew);
            const float p0    = __expf(v0 - mnew);
            const float p1    = __expf(v1 - mnew);
            float ps = p0 + p1;
            #pragma unroll
            for (int off = 1; off < 16; off <<= 1)
                ps += __shfl_xor(ps, off, 32);
            lrun[r] = lrun[r] * alpha + ps;
            mrun[r] = mnew;
            #pragma unroll
            for (int t = 0; t < 4; ++t) o[t][r] *= alpha;
            const int row = lane_hi * 8 + r;
            plds[wave][row][lane_lo]      = (_Float16)p0;
            plds[wave][row][16 + lane_lo] = (_Float16)p1;
        }

        // reload P in A-operand layout (same-wave LDS, in-order)
        V16H pa;
        pa.u[0] = *(const uint4*)&plds[wave][lane_lo][lane_hi * 8];
        pa.u[1] = *(const uint4*)&plds[wave][lane_lo][16 + lane_hi * 8];

        // O += P @ V   (V^T rows are contiguous in keys)
        #pragma unroll
        for (int t = 0; t < 4; ++t) {
            V16H vb;
            const _Float16* p = Vb + (size_t)(t * 16 + lane_lo) * Nn + k0 + lane_hi * 8;
            vb.u[0] = *(const uint4*)p;
            vb.u[1] = *(const uint4*)(p + 16);
            o[t] = __builtin_amdgcn_wmma_f32_16x16x32_f16(
                false, pa.v, false, vb.v, (short)0, o[t], false, false);
        }
    }

    // finalize: divide by softmax denominator, store concat layout [B][N][768]
    const int b = bh / Hh, h = bh - b * Hh;
    #pragma unroll
    for (int r = 0; r < 8; ++r) {
        const float inv = 1.0f / lrun[r];
        const int q = q0 + lane_hi * 8 + r;
        #pragma unroll
        for (int t = 0; t < 4; ++t) {
            const int d = t * 16 + lane_lo;
            attn[((size_t)(b * Nn + q)) * Dd + h * PD + d] = (_Float16)(o[t][r] * inv);
        }
    }
}

// ---------------------------------------------------------------------------
extern "C" void kernel_launch(void* const* d_in, const int* in_sizes, int n_in,
                              void* d_out, int out_size, void* d_ws, size_t ws_size,
                              hipStream_t stream) {
    (void)in_sizes; (void)n_in; (void)out_size; (void)ws_size;

    const float* x  = (const float*)d_in[0];
    const float* wq = (const float*)d_in[1];
    const float* bq = (const float*)d_in[2];
    const float* wk = (const float*)d_in[3];
    const float* bk = (const float*)d_in[4];
    const float* wv = (const float*)d_in[5];
    const float* bv = (const float*)d_in[6];
    const float* wo = (const float*)d_in[7];
    const float* bo = (const float*)d_in[8];

    char*  ws  = (char*)d_ws;
    size_t off = 0;
    auto alloc = [&](size_t bytes) {
        char* p = ws + off;
        off = (off + bytes + 255) & ~(size_t)255;
        return p;
    };

    _Float16* Xh  = (_Float16*)alloc((size_t)Mtot * Dd * 2);
    _Float16* WqT = (_Float16*)alloc((size_t)Dd * Dd * 2);
    _Float16* WkT = (_Float16*)alloc((size_t)Dd * Dd * 2);
    _Float16* WvT = (_Float16*)alloc((size_t)Dd * Dd * 2);
    _Float16* WoT = (_Float16*)alloc((size_t)Dd * Dd * 2);
    _Float16* Qh  = (_Float16*)alloc((size_t)Mtot * Dd * 2);
    _Float16* Kh  = (_Float16*)alloc((size_t)Mtot * Dd * 2);
    _Float16* Vt  = (_Float16*)alloc((size_t)Mtot * Dd * 2);
    _Float16* Ah  = (_Float16*)alloc((size_t)Mtot * Dd * 2);

    // 1) precision conversion / weight transposes
    cvt_f16_kernel<<<(Mtot * Dd) / 256, 256, 0, stream>>>(x, Xh, Mtot * Dd);
    transpose_w_f16_kernel<<<(Dd * Dd) / 256, 256, 0, stream>>>(wq, WqT);
    transpose_w_f16_kernel<<<(Dd * Dd) / 256, 256, 0, stream>>>(wk, WkT);
    transpose_w_f16_kernel<<<(Dd * Dd) / 256, 256, 0, stream>>>(wv, WvT);
    transpose_w_f16_kernel<<<(Dd * Dd) / 256, 256, 0, stream>>>(wo, WoT);

    // 2) QKV projections (Q carries the 1/sqrt(64) softmax scale)
    dim3 gg(Dd / 128, Mtot / 128);   // (6, 64)
    gemm_wmma_kernel<<<gg, 256, 0, stream>>>(Xh, WqT, bq, Qh, 0, 0.125f);
    gemm_wmma_kernel<<<gg, 256, 0, stream>>>(Xh, WkT, bk, Kh, 0, 1.0f);
    gemm_wmma_kernel<<<gg, 256, 0, stream>>>(Xh, WvT, bv, Vt, 1, 1.0f);

    // 3) flash attention per (b, h)
    attention_kernel<<<dim3(Nn / 128, Bb * Hh), 256, 0, stream>>>(Qh, Kh, Vt, Ah);

    // 4) output projection -> f32 d_out
    gemm_wmma_kernel<<<gg, 256, 0, stream>>>(Ah, WoT, bo, d_out, 2, 1.0f);
}
